// MultiHeadedAttention_70514773066567
// MI455X (gfx1250) — compile-verified, matrix operand order checked by reading, not run on hardware
//
#include <hip/hip_runtime.h>

// MultiHeadedAttention forward for gfx1250 (CDNA5, wave32, WMMA bf16).
// Pipeline: cvt(fp32->bf16) -> QKV GEMMs -> V transpose -> scores(QK^T, mask,
//           scale, skip fully-masked tiles) -> causal-aware row softmax
//           -> PV (causal-limited K loop) -> out projection.

#define D_MODEL  2048
#define S_LEN    2048
#define N_HEADS  8
#define HEAD_DIM 256
#define BATCH    2
#define SCALE_QK 0.0625f   // 256^-0.5
#define LDSK     40        // LDS row stride in u16 (80B, 16B aligned)

typedef __attribute__((ext_vector_type(16))) __bf16       v16bf;
typedef __attribute__((ext_vector_type(8)))  float        v8f;
typedef __attribute__((ext_vector_type(4)))  unsigned int uint4v;
typedef __attribute__((ext_vector_type(2)))  unsigned int uint2v;
typedef __attribute__((ext_vector_type(4)))  float        float4v;

union FragU { uint4v u[2]; v16bf v; };
union U4x8  { uint4v u; unsigned short s[8]; };

__device__ inline unsigned short f2bf(float f) {
  unsigned int u = __float_as_uint(f);
  unsigned int r = (u + 0x7FFFu + ((u >> 16) & 1u)) >> 16;  // RNE truncate
  return (unsigned short)r;
}

__device__ inline v8f zero8() {
  v8f z = {0.f, 0.f, 0.f, 0.f, 0.f, 0.f, 0.f, 0.f};
  return z;
}

// Load one 16x32 bf16 WMMA operand fragment from an LDS tile laid out
// row-major with stride LDSK. `row` already includes (lane & 15).
// Per-lane layout (ISA 7.12.2, 16-bit A/B): half = lane>>4;
//   vgprs 0..3 hold K = half*8 .. half*8+7, vgprs 4..7 hold K = 16+half*8 ..
__device__ inline v16bf load_frag(const unsigned short* lds, int row, int half) {
  const unsigned short* p = lds + row * LDSK + half * 8;
  FragU f;
  f.u[0] = *(const uint4v*)(p);
  f.u[1] = *(const uint4v*)(p + 16);
  return f.v;
}

// One K-step of the 128x128 block tile: 8 waves, wave (waveM 0..3, waveN 0..1)
// owns a 32x64 region = 2x4 WMMA 16x16x32 bf16 tiles.
__device__ inline void mma_8(const unsigned short* As, const unsigned short* Bs,
                             v8f acc[2][4], int waveM, int waveN, int lane) {
  const int l16 = lane & 15, half = lane >> 4;
  v16bf a[2], b[4];
#pragma unroll
  for (int mt = 0; mt < 2; ++mt)
    a[mt] = load_frag(As, waveM * 32 + mt * 16 + l16, half);
#pragma unroll
  for (int nt = 0; nt < 4; ++nt)
    b[nt] = load_frag(Bs, waveN * 64 + nt * 16 + l16, half);
#pragma unroll
  for (int mt = 0; mt < 2; ++mt)
#pragma unroll
    for (int nt = 0; nt < 4; ++nt)
      acc[mt][nt] = __builtin_amdgcn_wmma_f32_16x16x32_bf16(
          false, a[mt], false, b[nt], (short)0, acc[mt][nt], false, false);
}

// ---------------------------------------------------------------------------
// fp32 -> bf16 conversion (vectorized x4)
// ---------------------------------------------------------------------------
__global__ __launch_bounds__(256) void cvt_f32_bf16_kernel(
    const float* __restrict__ src, unsigned short* __restrict__ dst, int n4) {
  int i = blockIdx.x * 256 + threadIdx.x;
  if (i < n4) {
    float4v f = *(const float4v*)(src + (size_t)i * 4);
    uint2v pk;
    pk.x = (unsigned)f2bf(f.x) | ((unsigned)f2bf(f.y) << 16);
    pk.y = (unsigned)f2bf(f.z) | ((unsigned)f2bf(f.w) << 16);
    *(uint2v*)(dst + (size_t)i * 4) = pk;
  }
}

// ---------------------------------------------------------------------------
// Generic bf16 GEMM: C[M,N] = A[M,K] @ B[N,K]^T + bias, block tile 128x128.
// grid = (N/128, M/128), 256 threads. OUT_BF16 selects output precision.
// ---------------------------------------------------------------------------
template <bool OUT_BF16>
__global__ __launch_bounds__(256) void gemm_kernel(
    const unsigned short* __restrict__ A, int lda,
    const unsigned short* __restrict__ B, int ldb,
    const float* __restrict__ bias, void* __restrict__ Cout, int ldc, int K) {
  __shared__ unsigned short As[128 * LDSK];
  __shared__ unsigned short Bs[128 * LDSK];
  const int tid = threadIdx.x;
  const int mBase = blockIdx.y * 128, nBase = blockIdx.x * 128;
  const unsigned short* Ap = A + (size_t)mBase * lda;
  const unsigned short* Bp = B + (size_t)nBase * ldb;
  const int lane = tid & 31, wave = tid >> 5;
  const int waveM = wave & 3, waveN = wave >> 2;

  v8f acc[2][4];
#pragma unroll
  for (int mt = 0; mt < 2; ++mt)
#pragma unroll
    for (int nt = 0; nt < 4; ++nt) acc[mt][nt] = zero8();

  const int lr = tid >> 2;          // 0..63
  const int lc = (tid & 3) * 8;     // 0,8,16,24 (u16 units, 16B granules)

  for (int k0 = 0; k0 < K; k0 += 32) {
    __syncthreads();
#pragma unroll
    for (int p = 0; p < 2; ++p) {
      int r = lr + p * 64;
      *(uint4v*)(As + r * LDSK + lc) = *(const uint4v*)(Ap + (size_t)r * lda + k0 + lc);
      *(uint4v*)(Bs + r * LDSK + lc) = *(const uint4v*)(Bp + (size_t)r * ldb + k0 + lc);
    }
    if (k0 + 32 < K) {  // hint next K tile into cache (global_prefetch_b8)
      __builtin_prefetch(Ap + (size_t)lr * lda + k0 + 32 + lc, 0, 0);
      __builtin_prefetch(Bp + (size_t)lr * ldb + k0 + 32 + lc, 0, 0);
    }
    __syncthreads();
    mma_8(As, Bs, acc, waveM, waveN, lane);
  }

  const int l16 = lane & 15, rsel = lane >> 4;
#pragma unroll
  for (int nt = 0; nt < 4; ++nt) {
    int gn = nBase + waveN * 64 + nt * 16 + l16;
    float bv = bias ? bias[gn] : 0.f;
#pragma unroll
    for (int mt = 0; mt < 2; ++mt) {
#pragma unroll
      for (int v = 0; v < 8; ++v) {
        int gm = mBase + waveM * 32 + mt * 16 + rsel * 8 + v;
        float val = acc[mt][nt][v] + bv;
        if (OUT_BF16)
          ((unsigned short*)Cout)[(size_t)gm * ldc + gn] = f2bf(val);
        else
          ((float*)Cout)[(size_t)gm * ldc + gn] = val;
      }
    }
  }
}

// ---------------------------------------------------------------------------
// V transpose: Vt[bh, n, k] = V[b*S + k, h*hd + n]   (bf16, LDS 64x64 tiles)
// grid = (hd/64, S/64, B*H), 256 threads.
// ---------------------------------------------------------------------------
__global__ __launch_bounds__(256) void transpose_v_kernel(
    const unsigned short* __restrict__ V, unsigned short* __restrict__ Vt) {
  __shared__ unsigned short t[64 * 72];
  const int bh = blockIdx.z, b = bh >> 3, h = bh & 7;
  const int kBase = blockIdx.y * 64, nBase = blockIdx.x * 64;
  const int tid = threadIdx.x;
  const int r = tid >> 2, c = (tid & 3) * 16;

  const unsigned short* src =
      V + (size_t)(b * S_LEN + kBase + r) * D_MODEL + h * HEAD_DIM + nBase + c;
  *(uint4v*)(t + r * 72 + c)     = *(const uint4v*)(src);
  *(uint4v*)(t + r * 72 + c + 8) = *(const uint4v*)(src + 8);
  __syncthreads();

  U4x8 o0, o1;
#pragma unroll
  for (int j = 0; j < 8; ++j) {
    o0.s[j] = t[(c + j) * 72 + r];
    o1.s[j] = t[(c + 8 + j) * 72 + r];
  }
  unsigned short* dst =
      Vt + (size_t)(bh * HEAD_DIM + nBase + r) * S_LEN + kBase + c;
  *(uint4v*)(dst)     = o0.u;
  *(uint4v*)(dst + 8) = o1.u;
}

// ---------------------------------------------------------------------------
// Scores: P[bh, q, k] = scale * Q[bh,q,:] . K[bh,k,:]   (masked in-tile: -inf)
// Fully masked tiles are NOT written at all (softmax synthesizes them).
// grid = (S/128, S/128, B*H).
// ---------------------------------------------------------------------------
__global__ __launch_bounds__(256) void attn_scores_kernel(
    const unsigned short* __restrict__ Q, const unsigned short* __restrict__ Km,
    float* __restrict__ P) {
  const int bh = blockIdx.z, b = bh >> 3, h = bh & 7;
  const int qBase = blockIdx.y * 128, kBase = blockIdx.x * 128;
  if (kBase >= qBase + 128) return;  // entirely above the diagonal: no work

  __shared__ unsigned short As[128 * LDSK];
  __shared__ unsigned short Bs[128 * LDSK];
  const int tid = threadIdx.x;
  const unsigned short* Ap = Q + ((size_t)(b * S_LEN + qBase)) * D_MODEL + h * HEAD_DIM;
  const unsigned short* Bp = Km + ((size_t)(b * S_LEN + kBase)) * D_MODEL + h * HEAD_DIM;
  const int lane = tid & 31, wave = tid >> 5;
  const int waveM = wave & 3, waveN = wave >> 2;

  v8f acc[2][4];
#pragma unroll
  for (int mt = 0; mt < 2; ++mt)
#pragma unroll
    for (int nt = 0; nt < 4; ++nt) acc[mt][nt] = zero8();

  const int lr = tid >> 2, lc = (tid & 3) * 8;
  for (int k0 = 0; k0 < HEAD_DIM; k0 += 32) {
    __syncthreads();
#pragma unroll
    for (int p = 0; p < 2; ++p) {
      int r = lr + p * 64;
      *(uint4v*)(As + r * LDSK + lc) = *(const uint4v*)(Ap + (size_t)r * D_MODEL + k0 + lc);
      *(uint4v*)(Bs + r * LDSK + lc) = *(const uint4v*)(Bp + (size_t)r * D_MODEL + k0 + lc);
    }
    __syncthreads();
    mma_8(As, Bs, acc, waveM, waveN, lane);
  }

  const int l16 = lane & 15, rsel = lane >> 4;
#pragma unroll
  for (int nt = 0; nt < 4; ++nt) {
    int gk = kBase + waveN * 64 + nt * 16 + l16;
#pragma unroll
    for (int mt = 0; mt < 2; ++mt) {
#pragma unroll
      for (int v = 0; v < 8; ++v) {
        int gq = qBase + waveM * 32 + mt * 16 + rsel * 8 + v;
        float val = (gk <= gq) ? acc[mt][nt][v] * SCALE_QK : -__builtin_inff();
        P[((size_t)bh * S_LEN + gq) * S_LEN + gk] = val;
      }
    }
  }
}

// ---------------------------------------------------------------------------
// Causal-aware row softmax over probs [B*H*S rows x S].
// Masked positions (k > q) are never read from memory; all positions written.
// ---------------------------------------------------------------------------
__global__ __launch_bounds__(256) void softmax_kernel(float* __restrict__ P) {
  __shared__ float red[256];
  float* p = P + (size_t)blockIdx.x * S_LEN;
  const int q = (int)(blockIdx.x & (S_LEN - 1));  // row index within the head
  const int tid = threadIdx.x;
  float vals[8];
  float m = -__builtin_inff();
#pragma unroll
  for (int i = 0; i < 8; ++i) {
    int k = tid + i * 256;
    vals[i] = (k <= q) ? p[k] : -__builtin_inff();
    m = fmaxf(m, vals[i]);
  }
  red[tid] = m;
  __syncthreads();
  for (int s = 128; s > 0; s >>= 1) {
    if (tid < s) red[tid] = fmaxf(red[tid], red[tid + s]);
    __syncthreads();
  }
  float rowmax = red[0];
  __syncthreads();
  float sum = 0.f;
#pragma unroll
  for (int i = 0; i < 8; ++i) {
    vals[i] = __expf(vals[i] - rowmax);
    sum += vals[i];
  }
  red[tid] = sum;
  __syncthreads();
  for (int s = 128; s > 0; s >>= 1) {
    if (tid < s) red[tid] += red[tid + s];
    __syncthreads();
  }
  float inv = 1.f / red[0];
#pragma unroll
  for (int i = 0; i < 8; ++i) p[tid + i * 256] = vals[i] * inv;  // masked -> 0
}

// ---------------------------------------------------------------------------
// Context: Ctx[q, h*hd+n] = sum_k P[bh,q,k] * Vt[bh,n,k]  (bf16 out).
// Causal: only k-tiles with k0 < qBase+128 contribute (P is 0 beyond).
// P (fp32) converted+packed to bf16 into LDS; Vt loads are plain row-major.
// grid = (HEAD_DIM/128, S/128, B*H).
// ---------------------------------------------------------------------------
__global__ __launch_bounds__(256) void attn_pv_kernel(
    const float* __restrict__ P, const unsigned short* __restrict__ Vt,
    unsigned short* __restrict__ Ctx) {
  const int bh = blockIdx.z, b = bh >> 3, h = bh & 7;
  const int qBase = blockIdx.y * 128, nBase = blockIdx.x * 128;
  const int tid = threadIdx.x;

  __shared__ unsigned short As[128 * LDSK];   // P tile  [q][k] (bf16)
  __shared__ unsigned short Bs[128 * LDSK];   // Vt tile [n][k]
  const float* Pp = P + ((size_t)bh * S_LEN + qBase) * S_LEN;
  const unsigned short* Bp = Vt + (size_t)(bh * HEAD_DIM + nBase) * S_LEN;
  const int lane = tid & 31, wave = tid >> 5;
  const int waveM = wave & 3, waveN = wave >> 2;

  v8f acc[2][4];
#pragma unroll
  for (int mt = 0; mt < 2; ++mt)
#pragma unroll
    for (int nt = 0; nt < 4; ++nt) acc[mt][nt] = zero8();

  const int ar = tid >> 3, ac = (tid & 7) * 4;  // P loads: 32 rows/pass x4
  const int lr = tid >> 2, lc = (tid & 3) * 8;  // Vt loads (b128 granules)
  const int kEnd = qBase + 128;

  for (int k0 = 0; k0 < kEnd; k0 += 32) {
    __syncthreads();
#pragma unroll
    for (int p = 0; p < 4; ++p) {
      int r = ar + p * 32;
      float4v f = *(const float4v*)(Pp + (size_t)r * S_LEN + k0 + ac);
      uint2v pk;
      pk.x = (unsigned)f2bf(f.x) | ((unsigned)f2bf(f.y) << 16);
      pk.y = (unsigned)f2bf(f.z) | ((unsigned)f2bf(f.w) << 16);
      *(uint2v*)(As + r * LDSK + ac) = pk;
    }
#pragma unroll
    for (int p = 0; p < 2; ++p) {
      int r = lr + p * 64;
      *(uint4v*)(Bs + r * LDSK + lc) = *(const uint4v*)(Bp + (size_t)r * S_LEN + k0 + lc);
    }
    if (k0 + 32 < kEnd) {
      __builtin_prefetch(Pp + (size_t)ar * S_LEN + k0 + 32 + ac, 0, 0);
      __builtin_prefetch(Bp + (size_t)lr * S_LEN + k0 + 32 + lc, 0, 0);
    }
    __syncthreads();
    mma_8(As, Bs, acc, waveM, waveN, lane);
  }

  const int l16 = lane & 15, rsel = lane >> 4;
#pragma unroll
  for (int nt = 0; nt < 4; ++nt) {
    int gn = h * HEAD_DIM + nBase + waveN * 64 + nt * 16 + l16;
#pragma unroll
    for (int mt = 0; mt < 2; ++mt) {
#pragma unroll
      for (int v = 0; v < 8; ++v) {
        int gm = b * S_LEN + qBase + waveM * 32 + mt * 16 + rsel * 8 + v;
        Ctx[(size_t)gm * D_MODEL + gn] = f2bf(acc[mt][nt][v]);
      }
    }
  }
}

// ---------------------------------------------------------------------------
extern "C" void kernel_launch(void* const* d_in, const int* in_sizes, int n_in,
                              void* d_out, int out_size, void* d_ws, size_t ws_size,
                              hipStream_t stream) {
  (void)in_sizes; (void)n_in; (void)out_size; (void)ws_size;
  const float* x  = (const float*)d_in[0];
  const float* Wq = (const float*)d_in[1];
  const float* bq = (const float*)d_in[2];
  const float* Wk = (const float*)d_in[3];
  const float* bk = (const float*)d_in[4];
  const float* Wv = (const float*)d_in[5];
  const float* bv = (const float*)d_in[6];
  const float* Wo = (const float*)d_in[7];
  const float* bo = (const float*)d_in[8];

  const size_t nX = (size_t)BATCH * S_LEN * D_MODEL;  // 8,388,608
  const size_t nW = (size_t)D_MODEL * D_MODEL;        // 4,194,304

  unsigned short* xb  = (unsigned short*)d_ws;        // bf16 activations/weights
  unsigned short* wqb = xb + nX;
  unsigned short* wkb = wqb + nW;
  unsigned short* wvb = wkb + nW;
  unsigned short* wob = wvb + nW;
  unsigned short* qb  = wob + nW;
  unsigned short* kb  = qb + nX;
  unsigned short* vb  = kb + nX;
  unsigned short* ctx = vb + nX;
  unsigned short* vtb = ctx + nX;                     // total ws: 128 MiB

  float* outp  = (float*)d_out;                       // [B,S,D]
  float* probs = outp + nX;                           // [B,H,S,S]

  const dim3 blk(256);

  cvt_f32_bf16_kernel<<<dim3((unsigned)(nX / 4 / 256)), blk, 0, stream>>>(x,  xb,  (int)(nX / 4));
  cvt_f32_bf16_kernel<<<dim3((unsigned)(nW / 4 / 256)), blk, 0, stream>>>(Wq, wqb, (int)(nW / 4));
  cvt_f32_bf16_kernel<<<dim3((unsigned)(nW / 4 / 256)), blk, 0, stream>>>(Wk, wkb, (int)(nW / 4));
  cvt_f32_bf16_kernel<<<dim3((unsigned)(nW / 4 / 256)), blk, 0, stream>>>(Wv, wvb, (int)(nW / 4));
  cvt_f32_bf16_kernel<<<dim3((unsigned)(nW / 4 / 256)), blk, 0, stream>>>(Wo, wob, (int)(nW / 4));

  const dim3 ggrid(D_MODEL / 128, (BATCH * S_LEN) / 128);  // (16, 32)
  gemm_kernel<true ><<<ggrid, blk, 0, stream>>>(xb, D_MODEL, wqb, D_MODEL, bq, qb, D_MODEL, D_MODEL);
  gemm_kernel<true ><<<ggrid, blk, 0, stream>>>(xb, D_MODEL, wkb, D_MODEL, bk, kb, D_MODEL, D_MODEL);
  gemm_kernel<true ><<<ggrid, blk, 0, stream>>>(xb, D_MODEL, wvb, D_MODEL, bv, vb, D_MODEL, D_MODEL);

  transpose_v_kernel<<<dim3(HEAD_DIM / 64, S_LEN / 64, BATCH * N_HEADS), blk, 0, stream>>>(vb, vtb);

  attn_scores_kernel<<<dim3(S_LEN / 128, S_LEN / 128, BATCH * N_HEADS), blk, 0, stream>>>(qb, kb, probs);
  softmax_kernel<<<dim3(BATCH * N_HEADS * S_LEN), blk, 0, stream>>>(probs);
  attn_pv_kernel<<<dim3(HEAD_DIM / 128, S_LEN / 128, BATCH * N_HEADS), blk, 0, stream>>>(probs, vtb, ctx);

  gemm_kernel<false><<<ggrid, blk, 0, stream>>>(ctx, D_MODEL, wob, D_MODEL, bo, outp, D_MODEL, D_MODEL);
}
